// ParallelExperts_36730560315503
// MI455X (gfx1250) — matrix operand order repaired and checked by reading, not compiled, not run on hardware
//
#include <hip/hip_runtime.h>

// ---------------------------------------------------------------------------
// Grouped (MoE) GEMM for MI455X / gfx1250, wave32 + WMMA.
//   out[e*TPE + m, n] = sum_k X[e*TPE + m, k] * W[e, k, n]
// fp32 storage; compute via split-bf16 (hi+lo) -> 3x v_wmma_f32_16x16x32_bf16
// with fp32 accumulation (near-fp32 accuracy at 1/3 of bf16 matrix peak).
// ---------------------------------------------------------------------------

namespace pe {

constexpr int E     = 16;
constexpr int D_IN  = 1024;
constexpr int D_OUT = 4096;
constexpr int TPE   = 2048;        // tokens per expert

constexpr int BM = 128;            // block tile M
constexpr int BN = 128;            // block tile N
constexpr int BK = 32;             // K per WMMA step (bf16)
constexpr int LDT = BK + 8;        // padded LDS row stride (40 elems = 80B):
                                   // rows start in disjoint 4-bank groups
constexpr int MT = TPE / BM;       // 16 M tiles per expert
constexpr int NT = D_OUT / BN;     // 32 N tiles per expert
constexpr int NBLOCKS = E * MT * NT;

typedef __bf16 v16bf __attribute__((ext_vector_type(16)));
typedef __bf16 v8bf  __attribute__((ext_vector_type(8)));
typedef __bf16 v4bf  __attribute__((ext_vector_type(4)));
typedef float  v8f   __attribute__((ext_vector_type(8)));
typedef float  v4f   __attribute__((ext_vector_type(4)));

union FragBF16 { v16bf v; v8bf h[2]; };

} // namespace pe

using namespace pe;

__global__ __launch_bounds__(256)
void ParallelExperts_36730560315503_kernel(const float* __restrict__ X,
                                           const float* __restrict__ W,
                                           float* __restrict__ Y) {
  // LDS tiles: A (M-major, K contiguous) and B (N-major, K contiguous),
  // each split into bf16 hi / lo planes. 4 * 128 * 40 * 2B = 40 KB.
  __shared__ __bf16 Ah[BM * LDT];
  __shared__ __bf16 Al[BM * LDT];
  __shared__ __bf16 Bh[BN * LDT];
  __shared__ __bf16 Bl[BN * LDT];

  const int bid = blockIdx.x;
  const int e   = bid / (MT * NT);
  const int rem = bid % (MT * NT);
  const int mt  = rem / NT;
  const int nt  = rem % NT;
  const int m0  = mt * BM;
  const int n0  = nt * BN;

  const float* __restrict__ A = X + (size_t)(e * TPE + m0) * D_IN;          // BM x D_IN
  const float* __restrict__ B = W + (size_t)e * D_IN * D_OUT + n0;          // D_IN x BN
  float*       __restrict__ C = Y + (size_t)(e * TPE + m0) * D_OUT + n0;    // BM x BN

  const int t     = threadIdx.x;
  const int lane  = t & 31;
  const int wave  = t >> 5;            // 0..7
  const int wm    = wave & 3;          // 4 waves along M  (each 32 rows)
  const int wn    = wave >> 2;         // 2 waves along N  (each 64 cols)
  const int lrow  = lane & 15;
  const int lhalf = lane >> 4;         // 0 | 1

  // Per-wave accumulators: 2 (M) x 4 (N) tiles of 16x16 f32.
  v8f acc[2][4] = {};

  // B-fill mapping: each thread owns column n, 16 consecutive K values.
  const int bn = t & 127;
  const int bk = (t >> 7) * 16;        // 0 or 16

  for (int k0 = 0; k0 < D_IN; k0 += BK) {
    // ---- prefetch next K panel (global_prefetch_b8 path) ----
    if (k0 + BK < D_IN) {
      __builtin_prefetch(A + (size_t)(t >> 1) * D_IN + k0 + BK, 0, 1);
      __builtin_prefetch(B + (size_t)(k0 + BK + bk) * D_OUT + bn, 0, 1);
    }

    // ---- stage A tile: 128x32 fp32 -> bf16 hi/lo in LDS ----
#pragma unroll
    for (int i = 0; i < 4; ++i) {
      const int idx = i * 256 + t;           // 0..1023 float4 slots
      const int row = idx >> 3;              // 0..127
      const int kq  = idx & 7;               // float4 within row
      const v4f x = *(const v4f*)(A + (size_t)row * D_IN + k0 + kq * 4);
      v4bf hv, lv;
#pragma unroll
      for (int j = 0; j < 4; ++j) {
        const float xf = x[j];
        const __bf16 hi = (__bf16)xf;
        hv[j] = hi;
        lv[j] = (__bf16)(xf - (float)hi);
      }
      *(v4bf*)&Ah[row * LDT + kq * 4] = hv;
      *(v4bf*)&Al[row * LDT + kq * 4] = lv;
    }

    // ---- stage B tile: 32x128 fp32 (K-major in HBM) -> N-major bf16 in LDS ----
    {
      v8bf h0, h1, l0, l1;
#pragma unroll
      for (int j = 0; j < 8; ++j) {
        const float x = B[(size_t)(k0 + bk + j) * D_OUT + bn];
        const __bf16 hi = (__bf16)x;
        h0[j] = hi;
        l0[j] = (__bf16)(x - (float)hi);
      }
#pragma unroll
      for (int j = 0; j < 8; ++j) {
        const float x = B[(size_t)(k0 + bk + 8 + j) * D_OUT + bn];
        const __bf16 hi = (__bf16)x;
        h1[j] = hi;
        l1[j] = (__bf16)(x - (float)hi);
      }
      *(v8bf*)&Bh[bn * LDT + bk]     = h0;
      *(v8bf*)&Bh[bn * LDT + bk + 8] = h1;
      *(v8bf*)&Bl[bn * LDT + bk]     = l0;
      *(v8bf*)&Bl[bn * LDT + bk + 8] = l1;
    }

    __syncthreads();

    // ---- A fragments (ISA 16-bit A 16x32 layout):
    //  lanes 0-15:  M=lrow, K = 0..7  then 16..23
    //  lanes 16-31: M=lrow, K = 8..15 then 24..31
    FragBF16 afh[2], afl[2];
    const int akoff = lhalf * 8;
#pragma unroll
    for (int tm = 0; tm < 2; ++tm) {
      const int M = wm * 32 + tm * 16 + lrow;
      afh[tm].h[0] = *(const v8bf*)&Ah[M * LDT + akoff];
      afh[tm].h[1] = *(const v8bf*)&Ah[M * LDT + akoff + 16];
      afl[tm].h[0] = *(const v8bf*)&Al[M * LDT + akoff];
      afl[tm].h[1] = *(const v8bf*)&Al[M * LDT + akoff + 16];
    }

    // ---- B fragments: lane holds column N=lrow, K 0..7 then 16..23 (lanes 0-15)
    //      or K 8..15 then 24..31 (lanes 16-31) -> two b128 LDS reads.
    const int bks = lhalf * 8;
#pragma unroll
    for (int tn = 0; tn < 4; ++tn) {
      const int N = wn * 64 + tn * 16 + lrow;
      FragBF16 bh, bl;
      bh.h[0] = *(const v8bf*)&Bh[N * LDT + bks];
      bh.h[1] = *(const v8bf*)&Bh[N * LDT + bks + 16];
      bl.h[0] = *(const v8bf*)&Bl[N * LDT + bks];
      bl.h[1] = *(const v8bf*)&Bl[N * LDT + bks + 16];
#pragma unroll
      for (int tm = 0; tm < 2; ++tm) {
        // hi*hi + hi*lo + lo*hi, fp32 accumulate (drop lo*lo, ~2^-16 rel err)
        acc[tm][tn] = __builtin_amdgcn_wmma_f32_16x16x32_bf16(
            false, afh[tm].v, false, bh.v, (short)0, acc[tm][tn], false, false);
        acc[tm][tn] = __builtin_amdgcn_wmma_f32_16x16x32_bf16(
            false, afh[tm].v, false, bl.v, (short)0, acc[tm][tn], false, false);
        acc[tm][tn] = __builtin_amdgcn_wmma_f32_16x16x32_bf16(
            false, afl[tm].v, false, bh.v, (short)0, acc[tm][tn], false, false);
      }
    }

    __syncthreads();
  }

  // ---- epilogue: C/D 16x16 f32 layout: VGPR v -> M = v + 8*lhalf, N = lrow ----
#pragma unroll
  for (int tm = 0; tm < 2; ++tm) {
#pragma unroll
    for (int tn = 0; tn < 4; ++tn) {
      const int colg = wn * 64 + tn * 16 + lrow;
#pragma unroll
      for (int v = 0; v < 8; ++v) {
        const int rowg = wm * 32 + tm * 16 + lhalf * 8 + v;
        C[(size_t)rowg * D_OUT + colg] = acc[tm][tn][v];
      }
    }
  }
}

extern "C" void kernel_launch(void* const* d_in, const int* in_sizes, int n_in,
                              void* d_out, int out_size, void* d_ws, size_t ws_size,
                              hipStream_t stream) {
  (void)in_sizes; (void)n_in; (void)out_size; (void)d_ws; (void)ws_size;
  const float* X = (const float*)d_in[0];
  // d_in[1] = expert_size (int64, static uniform TPE per reference) -- unused
  const float* W = (const float*)d_in[2];
  float* Y = (float*)d_out;

  ParallelExperts_36730560315503_kernel<<<dim3(NBLOCKS), dim3(256), 0, stream>>>(X, W, Y);
}
